// AUAttnProcessor_16028817949037
// MI455X (gfx1250) — matrix operand order repaired and checked by reading, not compiled, not run on hardware
//
#include <hip/hip_runtime.h>
#include <hip/hip_bf16.h>

// AUAttnProcessor for MI455X (gfx1250): bf16 WMMA GEMMs + flash attention with
// async (memory->LDS) double-buffered K/V tile staging, ds_swizzle softmax
// reductions. B=4, S=2048, C=640, H=8, D=80 (padded to 96).

#define BB   4
#define SS   2048
#define CC   640
#define HH   8
#define DD   80
#define DP   96          // D padded to 3 * 32
#define BH   (BB*HH)
#define NAU_ 12

typedef __attribute__((ext_vector_type(16))) __bf16        v16bf;
typedef __attribute__((ext_vector_type(8)))  float         v8f;
typedef __attribute__((ext_vector_type(8)))  unsigned int  v8u;

__device__ __forceinline__ unsigned short f2bf(float x) {
  unsigned int u = __builtin_bit_cast(unsigned int, x);
  u += 0x7FFFu + ((u >> 16) & 1u);          // round-to-nearest-even
  return (unsigned short)(u >> 16);
}

__device__ __forceinline__ unsigned pk2bf(float lo, float hi) {
  return (unsigned)f2bf(lo) | ((unsigned)f2bf(hi) << 16);
}

// Lane-xor via DS_SWIZZLE_B32, group-of-32 mode: offset = xor_mask<<10 | and_mask(0x1f).
#define SWZ_XOR(x, m)                                                        \
  __builtin_bit_cast(float, __builtin_amdgcn_ds_swizzle(                     \
      __builtin_bit_cast(int, (x)), ((m) << 10) | 0x1F))

// Butterfly reductions over the 16-lane half (xor masks 1,2,4,8 keep lane&16).
__device__ __forceinline__ void rowmax8(float* v) {
#pragma unroll
  for (int g = 0; g < 8; ++g) v[g] = fmaxf(v[g], SWZ_XOR(v[g], 1));
#pragma unroll
  for (int g = 0; g < 8; ++g) v[g] = fmaxf(v[g], SWZ_XOR(v[g], 2));
#pragma unroll
  for (int g = 0; g < 8; ++g) v[g] = fmaxf(v[g], SWZ_XOR(v[g], 4));
#pragma unroll
  for (int g = 0; g < 8; ++g) v[g] = fmaxf(v[g], SWZ_XOR(v[g], 8));
}
__device__ __forceinline__ void rowsum8(float* v) {
#pragma unroll
  for (int g = 0; g < 8; ++g) v[g] += SWZ_XOR(v[g], 1);
#pragma unroll
  for (int g = 0; g < 8; ++g) v[g] += SWZ_XOR(v[g], 2);
#pragma unroll
  for (int g = 0; g < 8; ++g) v[g] += SWZ_XOR(v[g], 4);
#pragma unroll
  for (int g = 0; g < 8; ++g) v[g] += SWZ_XOR(v[g], 8);
}

// Async 16B copy global -> LDS (GV mode), tracked by ASYNCcnt.
__device__ __forceinline__ void async_copy16(unsigned lds_addr, const void* gaddr) {
  asm volatile("global_load_async_to_lds_b128 %0, %1, off"
               :: "v"(lds_addr), "v"(gaddr) : "memory");
}

#if __has_builtin(__builtin_amdgcn_s_wait_asynccnt)
#define WAIT_ASYNC(n) __builtin_amdgcn_s_wait_asynccnt(n)
#else
#define WAIT_ASYNC(n) asm volatile("s_wait_asynccnt %0" :: "i"(n) : "memory")
#endif

// A fragment (16xK tile, row-major [m][k] with 'stride' elements per row).
__device__ __forceinline__ v16bf frag_a(const unsigned short* t, int stride, int kbase, int lane) {
  int m   = lane & 15;
  int klo = kbase + ((lane & 16) ? 8 : 0);
  const unsigned int* pr = (const unsigned int*)(t + m * stride);
  int a = klo >> 1, b = (klo + 16) >> 1;
  v8u u;
  u[0] = pr[a];     u[1] = pr[a + 1]; u[2] = pr[a + 2]; u[3] = pr[a + 3];
  u[4] = pr[b];     u[5] = pr[b + 1]; u[6] = pr[b + 2]; u[7] = pr[b + 3];
  return __builtin_bit_cast(v16bf, u);
}

// B fragment from tile stored as [n][k] row-major ("transposed" B).
__device__ __forceinline__ v16bf frag_b(const unsigned short* t, int n0, int stride, int kbase, int lane) {
  int n  = n0 + (lane & 15);
  int kb = (kbase + ((lane & 16) ? 8 : 0)) >> 1;
  const unsigned int* pr = (const unsigned int*)(t + n * stride);
  v8u u;
#pragma unroll
  for (int i = 0; i < 4; ++i) { u[i] = pr[kb + i]; u[i + 4] = pr[kb + 8 + i]; }
  return __builtin_bit_cast(v16bf, u);
}

__device__ __forceinline__ v8f wmma_bf16(v16bf a, v16bf b, v8f c) {
  return __builtin_amdgcn_wmma_f32_16x16x32_bf16(false, a, false, b, (short)0, c, false, false);
}

// ---------------------------------------------------------------------------
// Kernel 1: Q/K/V projections, templated on Z. grid = (M/128, N/64).
// q: bf16 [bh][S][96], pre-scaled by SCALE*log2(e);  k: bf16 [bh][S][96];
// v: bf16 transposed [bh][80][S].
// ---------------------------------------------------------------------------
template <int Z>
__global__ __launch_bounds__(256) void proj_qkv_kernel(
    const float* __restrict__ hidden, const float* __restrict__ W,
    unsigned short* __restrict__ dst)
{
  __shared__ unsigned short ldsA[128 * 32];   // A tile [128][32] bf16
  __shared__ unsigned short ldsB[64 * 32];    // W tile transposed [n][k]
  const int tid = threadIdx.x, lane = tid & 31, wave = tid >> 5;
  const int row0 = blockIdx.x * 128;
  const int n0   = blockIdx.y * 64;

  v8f zero8 = {0, 0, 0, 0, 0, 0, 0, 0};
  v8f acc[4];
#pragma unroll
  for (int i = 0; i < 4; ++i) acc[i] = zero8;

  for (int k0 = 0; k0 < CC; k0 += 32) {
    __syncthreads();
#pragma unroll
    for (int j = 0; j < 8; ++j) {                        // 2048 f32 pairs of A
      int ep = j * 256 + tid;
      int r = ep >> 4, cp = ep & 15;
      float2 f = ((const float2*)(hidden + (size_t)(row0 + r) * CC + k0))[cp];
      ((unsigned int*)ldsA)[ep] = pk2bf(f.x, f.y);
    }
#pragma unroll
    for (int j = 0; j < 8; ++j) {                        // 32*64 B elems
      int e = j * 256 + tid;
      int k = e >> 6, n = e & 63;
      ldsB[n * 32 + k] = f2bf(W[(size_t)(k0 + k) * CC + n0 + n]);
    }
    __syncthreads();
    v16bf af = frag_a(ldsA + wave * 16 * 32, 32, 0, lane);
#pragma unroll
    for (int nt = 0; nt < 4; ++nt)
      acc[nt] = wmma_bf16(af, frag_b(ldsB, nt * 16, 32, 0, lane), acc[nt]);
  }

  const float QS = 0.11180339887498949f * 1.4426950408889634f; // SCALE*log2(e)
  const int ncol = lane & 15, moff = (lane & 16) ? 8 : 0;
#pragma unroll
  for (int nt = 0; nt < 4; ++nt) {
    int n = n0 + nt * 16 + ncol;
    int h = n / DD, d = n % DD;
#pragma unroll
    for (int g = 0; g < 8; ++g) {
      int r = row0 + wave * 16 + moff + g;
      int b = r >> 11, s = r & (SS - 1);
      float v = acc[nt][g];
      if (Z == 0)      dst[((size_t)((b * HH + h) * SS + s)) * DP + d] = f2bf(v * QS);
      else if (Z == 1) dst[((size_t)((b * HH + h) * SS + s)) * DP + d] = f2bf(v);
      else             dst[((size_t)((b * HH + h) * DD + d)) * SS + s] = f2bf(v);
    }
  }
}

// Zero the d in [80,96) pad columns of q and k (b32 pair stores).
__global__ __launch_bounds__(256) void pad_kernel(unsigned int* qb, unsigned int* kb) {
  size_t i = (size_t)blockIdx.x * 256 + threadIdx.x;   // over BH*S*8 uint pairs
  if (i >= (size_t)BH * SS * 8) return;
  size_t rs = i >> 3;
  int dp = (int)(i & 7);                               // pair index within pad
  qb[rs * (DP / 2) + 40 + dp] = 0;
  kb[rs * (DP / 2) + 40 + dp] = 0;
}

// AU key projection: auk bf16 [bh][16][96]; rows 13..15 and d-pads zero.
__global__ __launch_bounds__(256) void au_k_kernel(
    const float* __restrict__ au_emb, const float* __restrict__ null_tok,
    const float* __restrict__ Wak, unsigned short* __restrict__ auk)
{
  int i = blockIdx.x * 256 + threadIdx.x;
  if (i >= BH * 16 * DP) return;
  int dp = i % DP, key = (i / DP) & 15, bh = i / (DP * 16);
  int b = bh >> 3, h = bh & 7;
  float v = 0.f;
  if (dp < DD && key < 13) {
    const float* xr = (key < NAU_) ? (au_emb + (size_t)(b * NAU_ + key) * CC) : null_tok;
    int n = h * DD + dp;
    float s = 0.f;
    for (int k = 0; k < CC; ++k) s += xr[k] * Wak[(size_t)k * CC + n];
    v = s;
  }
  auk[i] = f2bf(v);
}

// AU value projection, transposed: auvt bf16 [bh][80][32]; null token (key 12)
// and pad keys zeroed per reference.
__global__ __launch_bounds__(256) void au_v_kernel(
    const float* __restrict__ au_emb, const float* __restrict__ Wav,
    unsigned short* __restrict__ auvt)
{
  int i = blockIdx.x * 256 + threadIdx.x;
  if (i >= BH * DD * 32) return;
  int key = i & 31, d = (i >> 5) % DD, bh = i / (32 * DD);
  int b = bh >> 3, h = bh & 7;
  float v = 0.f;
  if (key < NAU_) {
    const float* xr = au_emb + (size_t)(b * NAU_ + key) * CC;
    int n = h * DD + d;
    float s = 0.f;
    for (int k = 0; k < CC; ++k) s += xr[k] * Wav[(size_t)k * CC + n];
    v = s;
  }
  auvt[i] = f2bf(v);
}

// ---------------------------------------------------------------------------
// Kernel 2: flash attention + AU cross-attention + gamma fusion.
// grid = (S/128, B*H), 8 waves x 16 query rows. Double-buffered async K/V
// tile staging (ASYNCcnt). Writes bf16 pre [B][S][C].
// LDS map: buf{0,1}: [K 64x96 | Vt 80x64] at sel*22528; P (8x 16x64) at 45056.
// AU tables overlay buf0 before the main loop.
// ---------------------------------------------------------------------------
#define TILE_BYTES 22528   // 12288 (K) + 10240 (Vt)

__global__ __launch_bounds__(256) void attn_kernel(
    const unsigned short* __restrict__ qb, const unsigned short* __restrict__ kb,
    const unsigned short* __restrict__ vt, const unsigned short* __restrict__ auk,
    const unsigned short* __restrict__ auvt, const float* __restrict__ gamma,
    unsigned short* __restrict__ pre)
{
  __shared__ __align__(16) unsigned char smem[2 * TILE_BYTES + 16384];
  unsigned short* sAuk = (unsigned short*)(smem);            // [16][96] (AU phase)
  unsigned short* sAuv = (unsigned short*)(smem + 3072);     // [80][32] (AU phase)
  const int tid = threadIdx.x, lane = tid & 31, wave = tid >> 5;
  unsigned short* sP = (unsigned short*)(smem + 2 * TILE_BYTES) + wave * 16 * 64;
  const int bh = blockIdx.y, b = bh >> 3, h = bh & 7;
  const int q0 = blockIdx.x * 128;
  const int ncol = lane & 15, moff = (lane & 16) ? 8 : 0;
  const unsigned smem_base = (unsigned)(size_t)(&smem[0]);

  // Loop-invariant Q fragments straight from global (row-major, stride 96).
  const unsigned short* qrow = qb + (size_t)(bh * SS + q0 + wave * 16) * DP;
  v16bf qf[3];
#pragma unroll
  for (int ks = 0; ks < 3; ++ks) qf[ks] = frag_a(qrow, DP, ks * 32, lane);

  // ---- AU cross-attention phase (13 keys, loop-invariant) ----
  {
    const unsigned int* s0 = (const unsigned int*)(auk + (size_t)bh * 16 * DP);
    unsigned int* d0 = (unsigned int*)sAuk;
    for (int e = tid; e < 16 * DP / 2; e += 256) d0[e] = s0[e];
    const unsigned int* s1 = (const unsigned int*)(auvt + (size_t)bh * DD * 32);
    unsigned int* d1 = (unsigned int*)sAuv;
    for (int e = tid; e < DD * 32 / 2; e += 256) d1[e] = s1[e];
  }
  __syncthreads();

  v8f zero8 = {0, 0, 0, 0, 0, 0, 0, 0};
  v8f sau = zero8;
#pragma unroll
  for (int ks = 0; ks < 3; ++ks)
    sau = wmma_bf16(qf[ks], frag_b(sAuk, 0, DP, ks * 32, lane), sau);
  if (ncol >= 13) {                                    // mask pad keys 13..15
#pragma unroll
    for (int g = 0; g < 8; ++g) sau[g] = -1e30f;
  }
  float mau[8], pau[8], lau[8];
#pragma unroll
  for (int g = 0; g < 8; ++g) mau[g] = sau[g];
  rowmax8(mau);
#pragma unroll
  for (int g = 0; g < 8; ++g) { pau[g] = exp2f(sau[g] - mau[g]); lau[g] = pau[g]; }
  rowsum8(lau);
#pragma unroll
  for (int g = 0; g < 8; ++g) pau[g] /= lau[g];
#pragma unroll
  for (int g = 0; g < 8; ++g) {                        // stage P_au, zero cols 16..31
    int r = moff + g;
    sP[r * 64 + ncol]      = f2bf(pau[g]);
    sP[r * 64 + 16 + ncol] = 0;
  }
  v8f au_o[5];
#pragma unroll
  for (int dt = 0; dt < 5; ++dt) au_o[dt] = zero8;
  {
    v16bf pa = frag_a(sP, 64, 0, lane);
#pragma unroll
    for (int dt = 0; dt < 5; ++dt)
      au_o[dt] = wmma_bf16(pa, frag_b(sAuv, dt * 16, 32, 0, lane), au_o[dt]);
  }
  __syncthreads();   // AU LDS reads done; buf0 region may now be overwritten

  // ---- main flash-attention loop over 64-key tiles, double-buffered ----
  // issue_tile: 6 async b128 instructions per wave per tile (3 K + 3 V; uniform).
  const char* kbase0 = (const char*)(kb + (size_t)bh * SS * DP);
  const char* vbase0 = (const char*)(vt + (size_t)bh * DD * SS);
  auto issue_tile = [&](int t, int sel) {
    unsigned kdst = smem_base + (unsigned)(sel * TILE_BYTES);
    const char* kg = kbase0 + (size_t)t * (64 * DP * 2);
#pragma unroll
    for (int i = 0; i < 3; ++i) {                       // 768 x 16B chunks
      int e = i * 256 + tid;
      async_copy16(kdst + e * 16, kg + e * 16);
    }
    unsigned vdst = smem_base + (unsigned)(sel * TILE_BYTES + 12288);
    const char* vg = vbase0 + (size_t)t * 128;          // 64 keys * 2B
#pragma unroll
    for (int i = 0; i < 3; ++i) {                       // 640 x 16B chunks (tail dup)
      int e = (i < 2) ? (i * 256 + tid) : (512 + (tid & 127));
      int d = e >> 3, cu = e & 7;
      async_copy16(vdst + e * 16, vg + (size_t)d * (SS * 2) + cu * 16);
    }
  };

  v8f O[5];
#pragma unroll
  for (int dt = 0; dt < 5; ++dt) O[dt] = zero8;
  float mrun[8], lrun[8];
#pragma unroll
  for (int g = 0; g < 8; ++g) { mrun[g] = -1e30f; lrun[g] = 0.f; }

  issue_tile(0, 0);
  for (int t = 0; t < SS / 64; ++t) {
    if (t + 1 < SS / 64) { issue_tile(t + 1, (t + 1) & 1); WAIT_ASYNC(6); }
    else                 { WAIT_ASYNC(0); }
    __syncthreads();                                   // all waves' tile-t data landed
    const unsigned short* sK  = (const unsigned short*)(smem + (t & 1) * TILE_BYTES);
    const unsigned short* sVt = (const unsigned short*)(smem + (t & 1) * TILE_BYTES + 12288);

    v8f sc[4];
#pragma unroll
    for (int nt = 0; nt < 4; ++nt) sc[nt] = zero8;
#pragma unroll
    for (int ks = 0; ks < 3; ++ks)
#pragma unroll
      for (int nt = 0; nt < 4; ++nt)
        sc[nt] = wmma_bf16(qf[ks], frag_b(sK, nt * 16, DP, ks * 32, lane), sc[nt]);

    float tm[8];
#pragma unroll
    for (int g = 0; g < 8; ++g)
      tm[g] = fmaxf(fmaxf(sc[0][g], sc[1][g]), fmaxf(sc[2][g], sc[3][g]));
    rowmax8(tm);

    float alpha[8], ts[8];
#pragma unroll
    for (int g = 0; g < 8; ++g) {
      float mn = fmaxf(mrun[g], tm[g]);
      alpha[g] = exp2f(mrun[g] - mn);
      mrun[g]  = mn;
      ts[g]    = 0.f;
    }
    float pv[4][8];
#pragma unroll
    for (int nt = 0; nt < 4; ++nt)
#pragma unroll
      for (int g = 0; g < 8; ++g) {
        float p = exp2f(sc[nt][g] - mrun[g]);
        pv[nt][g] = p; ts[g] += p;
      }
    rowsum8(ts);
#pragma unroll
    for (int g = 0; g < 8; ++g) lrun[g] = lrun[g] * alpha[g] + ts[g];
#pragma unroll
    for (int dt = 0; dt < 5; ++dt)
#pragma unroll
      for (int g = 0; g < 8; ++g) O[dt][g] *= alpha[g];

    // stage P (C-layout -> A-fragment via per-wave LDS; same-wave DS ops in-order)
#pragma unroll
    for (int nt = 0; nt < 4; ++nt)
#pragma unroll
      for (int g = 0; g < 8; ++g)
        sP[(moff + g) * 64 + nt * 16 + ncol] = f2bf(pv[nt][g]);
#pragma unroll
    for (int kc = 0; kc < 2; ++kc) {
      v16bf pa = frag_a(sP, 64, kc * 32, lane);
#pragma unroll
      for (int dt = 0; dt < 5; ++dt)
        O[dt] = wmma_bf16(pa, frag_b(sVt, dt * 16, 64, kc * 32, lane), O[dt]);
    }
    __syncthreads();   // tile-t readers done before next iter's issue reuses buffer
  }

  // epilogue: main + gamma * au, to bf16 pre[B][S][C]
#pragma unroll
  for (int dt = 0; dt < 5; ++dt) {
#pragma unroll
    for (int g = 0; g < 8; ++g) {
      int srow = q0 + wave * 16 + moff + g;
      int d = dt * 16 + ncol;
      int c = h * DD + d;
      float val = O[dt][g] / lrun[g] + gamma[c] * au_o[dt][g];
      pre[((size_t)(b * SS + srow)) * CC + c] = f2bf(val);
    }
  }
}

// ---------------------------------------------------------------------------
// Kernel 3: out = pre @ Wo + bo + residual (fp32 output). grid = (64, 10).
// ---------------------------------------------------------------------------
__global__ __launch_bounds__(256) void out_proj_kernel(
    const unsigned short* __restrict__ pre, const float* __restrict__ Wo,
    const float* __restrict__ bo, const float* __restrict__ resid,
    float* __restrict__ out)
{
  __shared__ unsigned short ldsA[128 * 32];
  __shared__ unsigned short ldsB[64 * 32];
  const int tid = threadIdx.x, lane = tid & 31, wave = tid >> 5;
  const int row0 = blockIdx.x * 128, n0 = blockIdx.y * 64;

  v8f zero8 = {0, 0, 0, 0, 0, 0, 0, 0};
  v8f acc[4];
#pragma unroll
  for (int i = 0; i < 4; ++i) acc[i] = zero8;

  for (int k0 = 0; k0 < CC; k0 += 32) {
    __syncthreads();
#pragma unroll
    for (int j = 0; j < 8; ++j) {                        // 128*32 bf16 = 2048 uints
      int e = j * 256 + tid;
      int r = e >> 4, cu = e & 15;
      ((unsigned int*)ldsA)[e] =
          ((const unsigned int*)pre)[(((size_t)(row0 + r)) * CC + k0) / 2 + cu];
    }
#pragma unroll
    for (int j = 0; j < 8; ++j) {
      int e = j * 256 + tid;
      int k = e >> 6, n = e & 63;
      ldsB[n * 32 + k] = f2bf(Wo[(size_t)(k0 + k) * CC + n0 + n]);
    }
    __syncthreads();
    v16bf af = frag_a(ldsA + wave * 16 * 32, 32, 0, lane);
#pragma unroll
    for (int nt = 0; nt < 4; ++nt)
      acc[nt] = wmma_bf16(af, frag_b(ldsB, nt * 16, 32, 0, lane), acc[nt]);
  }

  const int ncol = lane & 15, moff = (lane & 16) ? 8 : 0;
#pragma unroll
  for (int nt = 0; nt < 4; ++nt) {
#pragma unroll
    for (int g = 0; g < 8; ++g) {
      int r = row0 + wave * 16 + moff + g;
      int n = n0 + nt * 16 + ncol;
      out[(size_t)r * CC + n] = acc[nt][g] + bo[n] + resid[(size_t)r * CC + n];
    }
  }
}

// ---------------------------------------------------------------------------
extern "C" void kernel_launch(void* const* d_in, const int* in_sizes, int n_in,
                              void* d_out, int out_size, void* d_ws, size_t ws_size,
                              hipStream_t stream) {
  (void)in_sizes; (void)n_in; (void)out_size; (void)ws_size;
  const float* hidden   = (const float*)d_in[0];
  const float* au_emb   = (const float*)d_in[1];
  const float* Wq       = (const float*)d_in[2];
  const float* Wk       = (const float*)d_in[3];
  const float* Wv       = (const float*)d_in[4];
  const float* Wak      = (const float*)d_in[5];
  const float* Wav      = (const float*)d_in[6];
  const float* null_tok = (const float*)d_in[7];
  const float* gamma    = (const float*)d_in[8];
  const float* Wo       = (const float*)d_in[9];
  const float* bo       = (const float*)d_in[10];
  float* out = (float*)d_out;

  unsigned char* ws = (unsigned char*)d_ws;
  size_t off = 0;
  auto alloc = [&](size_t bytes) {
    size_t p = off;
    off += (bytes + 255) & ~(size_t)255;
    return p;
  };
  unsigned short* qb   = (unsigned short*)(ws + alloc((size_t)BH * SS * DP * 2));
  unsigned short* kbuf = (unsigned short*)(ws + alloc((size_t)BH * SS * DP * 2));
  unsigned short* vt   = (unsigned short*)(ws + alloc((size_t)BH * DD * SS * 2));
  unsigned short* auk  = (unsigned short*)(ws + alloc((size_t)BH * 16 * DP * 2));
  unsigned short* auvt = (unsigned short*)(ws + alloc((size_t)BH * DD * 32 * 2));
  unsigned short* pre  = (unsigned short*)(ws + alloc((size_t)BB * SS * CC * 2));

  dim3 blk(256);
  proj_qkv_kernel<0><<<dim3(64, 10), blk, 0, stream>>>(hidden, Wq, qb);
  proj_qkv_kernel<1><<<dim3(64, 10), blk, 0, stream>>>(hidden, Wk, kbuf);
  proj_qkv_kernel<2><<<dim3(64, 10), blk, 0, stream>>>(hidden, Wv, vt);
  pad_kernel<<<dim3((BH * SS * 8) / 256), blk, 0, stream>>>((unsigned int*)qb, (unsigned int*)kbuf);
  au_k_kernel<<<dim3((BH * 16 * DP + 255) / 256), blk, 0, stream>>>(au_emb, null_tok, Wak, auk);
  au_v_kernel<<<dim3((BH * DD * 32 + 255) / 256), blk, 0, stream>>>(au_emb, Wav, auvt);
  attn_kernel<<<dim3(SS / 128, BH), blk, 0, stream>>>(qb, kbuf, vt, auk, auvt, gamma, pre);
  out_proj_kernel<<<dim3(64, 10), blk, 0, stream>>>(pre, Wo, bo, hidden, out);
}
